// GCNBranch_pos_34437047780013
// MI455X (gfx1250) — matrix operand order repaired and checked by reading, not compiled, not run on hardware
//
#include <hip/hip_runtime.h>

// ---------------------------------------------------------------------------
// GCN forward for MI455X (gfx1250, wave32, WMMA bf16).
//
// A_norm @ P is folded as D^-1/2 (A+I) (D^-1/2 P): the WMMA A-operand is
// exactly {0,1} (bf16-exact), P rows pre-scaled in fp32, fp32 accumulation.
//
// Full path (ws permitting): one fused pass converts A_pos -> (A+I) in bf16
// (128 MB, read 6x thereafter instead of 256 MB fp32) and computes dinv.
// P matrices are written transposed in bf16 by the producing GEMM epilogue,
// so the six propagation GEMMs stage both tiles as pure 16B row-major copies
// via GLOBAL_LOAD_ASYNC_TO_LDS_B128 (ASYNCcnt) when the builtin exists.
// ---------------------------------------------------------------------------

typedef __attribute__((ext_vector_type(16))) __bf16 v16bf;
typedef __attribute__((ext_vector_type(8)))  float  v8f;

union FragBF { v16bf v; unsigned int u[8]; };
union F4     { float4 v; float f[4]; };

__device__ __forceinline__ unsigned short f2bf(float x) {
  unsigned int u = __float_as_uint(x);
  u += 0x7FFFu + ((u >> 16) & 1u);          // round-to-nearest-even
  return (unsigned short)(u >> 16);
}

#define BM 64
#define BN 128
#define BK 64
#define LA 68   // LDS row stride (shorts), fp32-staged kernels
#define LB 68
#define LA2 72  // LDS row stride (shorts), async kernel: rows 16B-aligned
#define LB2 72

// -------- async copy global(bf16)->LDS, 16B per lane ------------------------
#if defined(__has_builtin)
#if __has_builtin(__builtin_amdgcn_global_load_async_to_lds_b128)
#define ASYNC_LDS 1
#endif
#endif
#ifndef ASYNC_LDS
#define ASYNC_LDS 0
#endif

#if ASYNC_LDS
typedef int v4i_ __attribute__((vector_size(16)));
typedef __attribute__((address_space(1))) v4i_* gas1_t;   // global b128
typedef __attribute__((address_space(3))) v4i_* las3_t;   // LDS b128
__device__ __forceinline__ void async_cp16(const void* g, void* l) {
  // generic->AS1: same address value; generic->AS3: low 32 bits are LDS offset
  __builtin_amdgcn_global_load_async_to_lds_b128(
      (gas1_t)(size_t)g, (las3_t)(size_t)(unsigned)(size_t)l, 0, 0);
}
#if __has_builtin(__builtin_amdgcn_s_wait_asynccnt)
__device__ __forceinline__ void wait_async() { __builtin_amdgcn_s_wait_asynccnt(0); }
#else
__device__ __forceinline__ void wait_async() { asm volatile("s_wait_asynccnt 0" ::: "memory"); }
#endif
#endif

// -------- shared WMMA phase: one 16x32 A frag x four 32x16 B frags ----------
__device__ __forceinline__ void mma_phase(
    const unsigned short* __restrict__ AsB, const unsigned short* __restrict__ BsB,
    int la, int lb, int wm, int wn, int half, int l16, int kk, v8f* acc)
{
  FragBF af;
  const unsigned int* Ap = (const unsigned int*)&AsB[(wm * 16 + l16) * la + kk];
  #pragma unroll
  for (int v = 0; v < 8; ++v)
    af.u[v] = Ap[(2 * v + (v >= 4 ? 8 : 0) + half * 8) >> 1];
  #pragma unroll
  for (int t = 0; t < 4; ++t) {
    FragBF bfr;
    const unsigned int* Bp =
        (const unsigned int*)&BsB[(wn * 64 + t * 16 + l16) * lb + kk];
    #pragma unroll
    for (int v = 0; v < 8; ++v)
      bfr.u[v] = Bp[(2 * v + (v >= 4 ? 8 : 0) + half * 8) >> 1];
    acc[t] = __builtin_amdgcn_wmma_f32_16x16x32_bf16(
        false, af.v, false, bfr.v, (short)0, acc[t], false, false);
  }
}

// -------- shared epilogue ---------------------------------------------------
__device__ __forceinline__ void epilogue(
    const v8f* acc, const float* bias, const float* base, const float* rowscale,
    float* C, unsigned short* outT, int ldt,
    int N, int NV, int ldc, int relu, float scale,
    int bm0, int bn0, int wm, int wn, int half, int l16)
{
  if (outT) {                     // bf16 transposed output (P^T producer)
    #pragma unroll
    for (int t = 0; t < 4; ++t) {
      int gn = bn0 + wn * 64 + t * 16 + l16;
      if (gn >= N) continue;
      int gm0 = bm0 + wm * 16 + half * 8;
      unsigned wd[4];
      #pragma unroll
      for (int p = 0; p < 4; ++p) {
        float v0 = acc[t][2 * p], v1 = acc[t][2 * p + 1];
        if (rowscale) { v0 *= rowscale[gm0 + 2 * p]; v1 *= rowscale[gm0 + 2 * p + 1]; }
        v0 *= scale; v1 *= scale;
        wd[p] = (unsigned)f2bf(v0) | ((unsigned)f2bf(v1) << 16);
      }
      uint4 w; w.x = wd[0]; w.y = wd[1]; w.z = wd[2]; w.w = wd[3];
      *(uint4*)&outT[(size_t)gn * ldt + gm0] = w;    // 16B store, aligned
    }
    return;
  }
  #pragma unroll
  for (int t = 0; t < 4; ++t) {
    int gn = bn0 + wn * 64 + t * 16 + l16;
    if (gn >= N) continue;
    float bv = (bias && gn < NV) ? bias[gn] : 0.f;
    #pragma unroll
    for (int r = 0; r < 8; ++r) {
      int gm = bm0 + wm * 16 + half * 8 + r;
      float v = acc[t][r];
      if (rowscale) v *= rowscale[gm];
      v += bv;
      if (relu) v = fmaxf(v, 0.f);
      v *= scale;
      size_t idx = (size_t)gm * ldc + gn;
      if (base) v += base[idx];
      C[idx] = v;
    }
  }
}

// ---------------- fast fp32-input kernel: no guards, double buffered --------
__launch_bounds__(256, 1)
__global__ void wmma_gemm_fast(
    const float* __restrict__ A, const float* __restrict__ B,
    const float* __restrict__ bias, const float* __restrict__ base,
    const float* __restrict__ rowscale, float* __restrict__ C,
    unsigned short* __restrict__ outT, int ldt,
    int N, int NV, int Kt, int lda, int ldb, int ldc,
    int adj, int relu, float scale)
{
  __shared__ unsigned short As[2][BM * LA];
  __shared__ unsigned short Bs[2][BN * LB];

  const int tid = threadIdx.x, lane = tid & 31, wave = tid >> 5;
  const int wm = wave & 3, wn = wave >> 2;
  const int half = lane >> 4, l16 = lane & 15;
  const int bm0 = blockIdx.y * BM, bn0 = blockIdx.x * BN;

  const int aM = tid >> 4, aK = (tid & 15) * 4;
  const int bK = tid >> 5, bN = (tid & 31) * 4;

  F4 aReg[4];
  F4 bReg[8];

  auto loadTiles = [&](int ks) {
    const float* Ag = A + (size_t)(bm0 + aM) * lda + (ks + aK);
    #pragma unroll
    for (int i = 0; i < 4; ++i)
      aReg[i].v = *(const float4*)(Ag + (size_t)(16 * i) * lda);
    const float* Bg = B + (size_t)(ks + bK) * ldb + (bn0 + bN);
    #pragma unroll
    for (int i = 0; i < 8; ++i)
      bReg[i].v = *(const float4*)(Bg + (size_t)(8 * i) * ldb);
  };

  auto storeTiles = [&](int nbuf, int ks) {
    const bool dg = (adj != 0) && (ks == bm0);
    #pragma unroll
    for (int i = 0; i < 4; ++i) {
      int m = aM + 16 * i;
      float f0 = aReg[i].f[0], f1 = aReg[i].f[1];
      float f2 = aReg[i].f[2], f3 = aReg[i].f[3];
      if (dg) {
        if (m == aK + 0) f0 += 1.f;
        if (m == aK + 1) f1 += 1.f;
        if (m == aK + 2) f2 += 1.f;
        if (m == aK + 3) f3 += 1.f;
      }
      uint2 p;
      p.x = (unsigned)f2bf(f0) | ((unsigned)f2bf(f1) << 16);
      p.y = (unsigned)f2bf(f2) | ((unsigned)f2bf(f3) << 16);
      *(uint2*)&As[nbuf][m * LA + aK] = p;
    }
    #pragma unroll
    for (int i = 0; i < 8; ++i) {
      int k = bK + 8 * i;
      #pragma unroll
      for (int j = 0; j < 4; ++j)
        Bs[nbuf][(bN + j) * LB + k] = f2bf(bReg[i].f[j]);
    }
  };

  v8f acc[4] = {};
  loadTiles(0);
  storeTiles(0, 0);
  __syncthreads();

  int buf = 0;
  for (int k0 = 0; k0 < Kt; k0 += BK) {
    const bool more = (k0 + BK) < Kt;
    if (more) loadTiles(k0 + BK);
    #pragma unroll
    for (int kk = 0; kk < BK; kk += 32)
      mma_phase(As[buf], Bs[buf], LA, LB, wm, wn, half, l16, kk, acc);
    if (more) storeTiles(buf ^ 1, k0 + BK);
    __syncthreads();
    buf ^= 1;
  }
  epilogue(acc, bias, base, rowscale, C, outT, ldt,
           N, NV, ldc, relu, scale, bm0, bn0, wm, wn, half, l16);
}

// ---------------- guarded fp32 kernel (small GEMMs with 62-wide weights) ----
__launch_bounds__(256, 1)
__global__ void wmma_gemm_guarded(
    const float* __restrict__ A, const float* __restrict__ B,
    const float* __restrict__ bias, const float* __restrict__ base,
    const float* __restrict__ rowscale, float* __restrict__ C,
    unsigned short* __restrict__ outT, int ldt,
    int N, int NV, int K, int Kt, int lda, int ldb, int ldc,
    int relu, float scale)
{
  __shared__ unsigned short As[BM * LA];
  __shared__ unsigned short Bs[BN * LB];

  const int tid = threadIdx.x, lane = tid & 31, wave = tid >> 5;
  const int wm = wave & 3, wn = wave >> 2;
  const int half = lane >> 4, l16 = lane & 15;
  const int bm0 = blockIdx.y * BM, bn0 = blockIdx.x * BN;

  v8f acc[4] = {};
  for (int k0 = 0; k0 < Kt; k0 += BK) {
    #pragma unroll
    for (int i = 0; i < (BM * BK) / 256; ++i) {
      int e = tid + i * 256;
      int m = e >> 6, k = e & 63;
      As[m * LA + k] = f2bf(A[(size_t)(bm0 + m) * lda + k0 + k]);
    }
    #pragma unroll
    for (int i = 0; i < (BN * BK) / 256; ++i) {
      int e = tid + i * 256;
      int n = e >> 6, k = e & 63;
      int gn = bn0 + n, gk = k0 + k;
      float b = (gk < K && gn < NV) ? B[(size_t)gk * ldb + gn] : 0.f;
      Bs[n * LB + k] = f2bf(b);
    }
    __syncthreads();
    #pragma unroll
    for (int kk = 0; kk < BK; kk += 32)
      mma_phase(As, Bs, LA, LB, wm, wn, half, l16, kk, acc);
    __syncthreads();
  }
  epilogue(acc, bias, base, rowscale, C, outT, ldt,
           N, NV, ldc, relu, scale, bm0, bn0, wm, wn, half, l16);
}

// ---------------- propagation kernel: bf16 inputs, async LDS staging --------
__launch_bounds__(256, 1)
__global__ void wmma_gemm_adj(
    const unsigned short* __restrict__ Ah,   // (A+I) bf16, ld = ldA
    const unsigned short* __restrict__ Bt,   // P^T bf16, ld = ldB
    const float* __restrict__ bias, const float* __restrict__ base,
    const float* __restrict__ rowscale, float* __restrict__ C,
    int N, int NV, int Kt, int ldA, int ldB, int ldc,
    int relu, float scale)
{
  __shared__ unsigned short As[2][BM * LA2];
  __shared__ unsigned short Bs[2][BN * LB2];

  const int tid = threadIdx.x, lane = tid & 31, wave = tid >> 5;
  const int wm = wave & 3, wn = wave >> 2;
  const int half = lane >> 4, l16 = lane & 15;
  const int bm0 = blockIdx.y * BM, bn0 = blockIdx.x * BN;

  // 16B copy units: row = cr (+32i), 16B segment cs within 128B row
  const int cr = tid >> 3;
  const int cs = (tid & 7) * 8;

#if ASYNC_LDS
  auto issueTiles = [&](int ks, int nbuf) {
    #pragma unroll
    for (int i = 0; i < 2; ++i) {
      int m = cr + 32 * i;
      async_cp16(Ah + (size_t)(bm0 + m) * ldA + ks + cs, &As[nbuf][m * LA2 + cs]);
    }
    #pragma unroll
    for (int i = 0; i < 4; ++i) {
      int n = cr + 32 * i;
      async_cp16(Bt + (size_t)(bn0 + n) * ldB + ks + cs, &Bs[nbuf][n * LB2 + cs]);
    }
  };
#else
  uint4 aR[2], bR[4];
  auto loadR = [&](int ks) {
    #pragma unroll
    for (int i = 0; i < 2; ++i)
      aR[i] = *(const uint4*)(Ah + (size_t)(bm0 + cr + 32 * i) * ldA + ks + cs);
    #pragma unroll
    for (int i = 0; i < 4; ++i)
      bR[i] = *(const uint4*)(Bt + (size_t)(bn0 + cr + 32 * i) * ldB + ks + cs);
  };
  auto storeR = [&](int nbuf) {
    #pragma unroll
    for (int i = 0; i < 2; ++i)
      *(uint4*)&As[nbuf][(cr + 32 * i) * LA2 + cs] = aR[i];
    #pragma unroll
    for (int i = 0; i < 4; ++i)
      *(uint4*)&Bs[nbuf][(cr + 32 * i) * LB2 + cs] = bR[i];
  };
#endif

  v8f acc[4] = {};
#if ASYNC_LDS
  issueTiles(0, 0);
  wait_async();
#else
  loadR(0);
  storeR(0);
#endif
  __syncthreads();

  int buf = 0;
  for (int k0 = 0; k0 < Kt; k0 += BK) {
    const bool more = (k0 + BK) < Kt;
#if ASYNC_LDS
    if (more) issueTiles(k0 + BK, buf ^ 1);    // DMA into other buffer
#else
    if (more) loadR(k0 + BK);
#endif
    #pragma unroll
    for (int kk = 0; kk < BK; kk += 32)
      mma_phase(As[buf], Bs[buf], LA2, LB2, wm, wn, half, l16, kk, acc);
#if ASYNC_LDS
    if (more) wait_async();
#else
    if (more) storeR(buf ^ 1);
#endif
    __syncthreads();
    buf ^= 1;
  }
  epilogue(acc, bias, base, rowscale, C, nullptr, 0,
           N, NV, ldc, relu, scale, bm0, bn0, wm, wn, half, l16);
}

// ---------------- fused pass: Ahat = (A+I) bf16, dinv = rsqrt(rowsum+1) -----
__launch_bounds__(256)
__global__ void prep_adj_kernel(const float* __restrict__ A,
                                unsigned short* __restrict__ Ah,
                                float* __restrict__ dinv, int n)
{
  __shared__ float red[256];
  const int r = blockIdx.x;
  const float* Ar = A + (size_t)r * n;
  float s = 0.f;
  #pragma unroll
  for (int i = 0; i < 4; ++i) {                 // n/(8*256) = 4
    int col = (threadIdx.x + i * 256) * 8;
    F4 a, b;
    a.v = *(const float4*)(Ar + col);
    b.v = *(const float4*)(Ar + col + 4);
    s += (a.f[0] + a.f[1]) + (a.f[2] + a.f[3]) +
         (b.f[0] + b.f[1]) + (b.f[2] + b.f[3]);
    #pragma unroll
    for (int j = 0; j < 4; ++j) if (col + j == r) a.f[j] += 1.f;
    #pragma unroll
    for (int j = 0; j < 4; ++j) if (col + 4 + j == r) b.f[j] += 1.f;
    uint4 w;
    w.x = (unsigned)f2bf(a.f[0]) | ((unsigned)f2bf(a.f[1]) << 16);
    w.y = (unsigned)f2bf(a.f[2]) | ((unsigned)f2bf(a.f[3]) << 16);
    w.z = (unsigned)f2bf(b.f[0]) | ((unsigned)f2bf(b.f[1]) << 16);
    w.w = (unsigned)f2bf(b.f[2]) | ((unsigned)f2bf(b.f[3]) << 16);
    *(uint4*)&Ah[(size_t)r * n + col] = w;
  }
  red[threadIdx.x] = s;
  __syncthreads();
  for (int off = 128; off > 0; off >>= 1) {
    if ((int)threadIdx.x < off) red[threadIdx.x] += red[threadIdx.x + off];
    __syncthreads();
  }
  if (threadIdx.x == 0) {
    float d = red[0] + 1.0f;
    dinv[r] = d > 0.f ? rsqrtf(d) : 0.f;
  }
}

// dinv only (fallback path)
__launch_bounds__(256)
__global__ void degree_kernel(const float* __restrict__ A,
                              float* __restrict__ dinv, int n)
{
  __shared__ float red[256];
  const int row = blockIdx.x;
  const float4* Ar = (const float4*)(A + (size_t)row * n);
  float s = 0.f;
  const int n4 = n >> 2;
  for (int j = threadIdx.x; j < n4; j += blockDim.x) {
    float4 v = Ar[j];
    s += (v.x + v.y) + (v.z + v.w);
  }
  red[threadIdx.x] = s;
  __syncthreads();
  for (int off = 128; off > 0; off >>= 1) {
    if ((int)threadIdx.x < off) red[threadIdx.x] += red[threadIdx.x + off];
    __syncthreads();
  }
  if (threadIdx.x == 0) {
    float d = red[0] + 1.0f;
    dinv[row] = d > 0.f ? rsqrtf(d) : 0.f;
  }
}

// ---------------- launch helpers -------------------------------------------
static void gfast(hipStream_t s, const float* A, const float* B,
                  const float* bias, const float* base, const float* rs,
                  float* C, unsigned short* outT, int ldt,
                  int M, int N, int NV, int Kt,
                  int lda, int ldb, int ldc, int adj, int relu, float scale)
{
  dim3 grid((N + BN - 1) / BN, M / BM);
  wmma_gemm_fast<<<grid, 256, 0, s>>>(A, B, bias, base, rs, C, outT, ldt,
                                      N, NV, Kt, lda, ldb, ldc, adj, relu, scale);
}

static void gguard(hipStream_t s, const float* A, const float* B,
                   const float* bias, const float* base, const float* rs,
                   float* C, unsigned short* outT, int ldt,
                   int M, int N, int NV, int K, int Kt,
                   int lda, int ldb, int ldc, int relu, float scale)
{
  dim3 grid((N + BN - 1) / BN, M / BM);
  wmma_gemm_guarded<<<grid, 256, 0, s>>>(A, B, bias, base, rs, C, outT, ldt,
                                         N, NV, K, Kt, lda, ldb, ldc, relu, scale);
}

static void gadj(hipStream_t s, const unsigned short* Ah, const unsigned short* Bt,
                 const float* bias, const float* base, const float* rs, float* C,
                 int M, int N, int NV, int Kt, int ldA, int ldB, int ldc,
                 int relu, float scale)
{
  dim3 grid((N + BN - 1) / BN, M / BM);
  wmma_gemm_adj<<<grid, 256, 0, s>>>(Ah, Bt, bias, base, rs, C,
                                     N, NV, Kt, ldA, ldB, ldc, relu, scale);
}

extern "C" void kernel_launch(void* const* d_in, const int* in_sizes, int n_in,
                              void* d_out, int out_size, void* d_ws, size_t ws_size,
                              hipStream_t stream)
{
  (void)in_sizes; (void)n_in; (void)out_size;
  constexpr int NN = 8192, IN = 1024, F = 256, MID = 62, MIDP = 64;

  const float* x    = (const float*)d_in[0];
  const float* Apos = (const float*)d_in[1];
  const float* W1 = (const float*)d_in[2];  const float* b1  = (const float*)d_in[3];
  const float* W2 = (const float*)d_in[4];  const float* b2  = (const float*)d_in[5];
  const float* W3 = (const float*)d_in[6];  const float* b3  = (const float*)d_in[7];
  const float* G1 = (const float*)d_in[8];  const float* gb1 = (const float*)d_in[9];
  const float* G2 = (const float*)d_in[10]; const float* gb2 = (const float*)d_in[11];
  const float* G3 = (const float*)d_in[12]; const float* gb3 = (const float*)d_in[13];
  const float* G4 = (const float*)d_in[14]; const float* gb4 = (const float*)d_in[15];
  const float* G5 = (const float*)d_in[16]; const float* gb5 = (const float*)d_in[17];
  const float* G6 = (const float*)d_in[18]; const float* gb6 = (const float*)d_in[19];

  float* ws   = (float*)d_ws;
  float* dinv = ws;                                   // NN floats
  float* L0   = dinv + NN;                            // NN*F fp32
  float* L1   = L0 + (size_t)NN * F;                  // NN*F fp32
  const size_t baseF = (size_t)NN + 2 * (size_t)NN * F;
  float*          Pf = ws + baseF;                    // fallback: NN*F fp32
  unsigned short* PT = (unsigned short*)(ws + baseF); // full: F x NN bf16
  unsigned short* Ah = PT + (size_t)F * NN;           // full: NN x NN bf16
  float* out = (float*)d_out;

  const size_t fullBytes = baseF * sizeof(float) +
      ((size_t)F * NN + (size_t)NN * NN) * sizeof(unsigned short);

  if (ws_size >= fullBytes) {
    // ---------------- bf16 pipeline ----------------
    prep_adj_kernel<<<NN, 256, 0, stream>>>(Apos, Ah, dinv, NN);

    // x1_lin = x @ W1 + b1                                  -> L0 fp32
    gfast(stream, x, W1, b1, nullptr, nullptr, L0, nullptr, 0,
          NN, F, F, IN, IN, F, F, 0, 0, 1.f);
    // P1^T = (dinv ⊙ (x1_lin @ G1))^T  bf16                 -> PT
    gfast(stream, L0, G1, nullptr, nullptr, dinv, nullptr, PT, NN,
          NN, F, F, F, F, F, 0, 0, 0, 1.f);
    // x1 = x1_lin + relu(dinv ⊙ (Ahat @ P1) + gb1)          -> L1
    gadj(stream, Ah, PT, gb1, L0, dinv, L1, NN, F, F, NN, NN, NN, F, 1, 1.f);

    // x2_lin (62 -> pad 64, zero pads)                      -> L0 (ld 64)
    gguard(stream, L1, W2, b2, nullptr, nullptr, L0, nullptr, 0,
           NN, MIDP, MID, F, F, F, MID, MIDP, 0, 1.f);
    // P2^T bf16 (rows 62,63 zero)                           -> PT
    gguard(stream, L0, G2, nullptr, nullptr, dinv, nullptr, PT, NN,
           NN, MIDP, MID, MID, MIDP, MIDP, MID, 0, 0, 1.f);
    // x2 = x2_lin + relu(gcn)                               -> L1 (ld 64)
    gadj(stream, Ah, PT, gb2, L0, dinv, L1, NN, MIDP, MID, NN, NN, NN, MIDP, 1, 1.f);

    // x3_lin = x2 @ W3 + b3                                 -> L0 (ld 256)
    gguard(stream, L1, W3, b3, nullptr, nullptr, L0, nullptr, 0,
           NN, F, F, MID, MIDP, MIDP, F, F, 0, 1.f);
    // P3^T                                                  -> PT
    gfast(stream, L0, G3, nullptr, nullptr, dinv, nullptr, PT, NN,
          NN, F, F, F, F, F, 0, 0, 0, 1.f);
    // x3 = x3_lin + 0.5*relu(gcn)                           -> L1
    gadj(stream, Ah, PT, gb3, L0, dinv, L1, NN, F, F, NN, NN, NN, F, 1, 0.5f);

    // P4^T; x4 = x3 + 0.5*relu(gcn)                         -> L0
    gfast(stream, L1, G4, nullptr, nullptr, dinv, nullptr, PT, NN,
          NN, F, F, F, F, F, 0, 0, 0, 1.f);
    gadj(stream, Ah, PT, gb4, L1, dinv, L0, NN, F, F, NN, NN, NN, F, 1, 0.5f);

    // P5^T; x5 = x4 + 0.25*relu(gcn)                        -> L1
    gfast(stream, L0, G5, nullptr, nullptr, dinv, nullptr, PT, NN,
          NN, F, F, F, F, F, 0, 0, 0, 1.f);
    gadj(stream, Ah, PT, gb5, L0, dinv, L1, NN, F, F, NN, NN, NN, F, 1, 0.25f);

    // P6^T; x6 = x5 + 0.25*gcn (no relu)                    -> out
    gfast(stream, L1, G6, nullptr, nullptr, dinv, nullptr, PT, NN,
          NN, F, F, F, F, F, 0, 0, 0, 1.f);
    gadj(stream, Ah, PT, gb6, L1, dinv, out, NN, F, F, NN, NN, NN, F, 0, 0.25f);
  } else {
    // ---------------- fp32 fallback pipeline ----------------
    degree_kernel<<<NN, 256, 0, stream>>>(Apos, dinv, NN);

    gfast(stream, x, W1, b1, nullptr, nullptr, L0, nullptr, 0,
          NN, F, F, IN, IN, F, F, 0, 0, 1.f);
    gfast(stream, L0, G1, nullptr, nullptr, dinv, Pf, nullptr, 0,
          NN, F, F, F, F, F, F, 0, 0, 1.f);
    gfast(stream, Apos, Pf, gb1, L0, dinv, L1, nullptr, 0,
          NN, F, F, NN, NN, F, F, 1, 1, 1.f);

    gguard(stream, L1, W2, b2, nullptr, nullptr, L0, nullptr, 0,
           NN, MIDP, MID, F, F, F, MID, MIDP, 0, 1.f);
    gguard(stream, L0, G2, nullptr, nullptr, dinv, Pf, nullptr, 0,
           NN, MIDP, MID, MID, MIDP, MIDP, MID, MIDP, 0, 1.f);
    gfast(stream, Apos, Pf, gb2, L0, dinv, L1, nullptr, 0,
          NN, MIDP, MID, NN, NN, MIDP, MIDP, 1, 1, 1.f);

    gguard(stream, L1, W3, b3, nullptr, nullptr, L0, nullptr, 0,
           NN, F, F, MID, MIDP, MIDP, F, F, 0, 1.f);
    gfast(stream, L0, G3, nullptr, nullptr, dinv, Pf, nullptr, 0,
          NN, F, F, F, F, F, F, 0, 0, 1.f);
    gfast(stream, Apos, Pf, gb3, L0, dinv, L1, nullptr, 0,
          NN, F, F, NN, NN, F, F, 1, 1, 0.5f);

    gfast(stream, L1, G4, nullptr, nullptr, dinv, Pf, nullptr, 0,
          NN, F, F, F, F, F, F, 0, 0, 1.f);
    gfast(stream, Apos, Pf, gb4, L1, dinv, L0, nullptr, 0,
          NN, F, F, NN, NN, F, F, 1, 1, 0.5f);

    gfast(stream, L0, G5, nullptr, nullptr, dinv, Pf, nullptr, 0,
          NN, F, F, F, F, F, F, 0, 0, 1.f);
    gfast(stream, Apos, Pf, gb5, L0, dinv, L1, nullptr, 0,
          NN, F, F, NN, NN, F, F, 1, 1, 0.25f);

    gfast(stream, L1, G6, nullptr, nullptr, dinv, Pf, nullptr, 0,
          NN, F, F, F, F, F, F, 0, 0, 1.f);
    gfast(stream, Apos, Pf, gb6, L1, dinv, out, nullptr, 0,
          NN, F, F, NN, NN, F, F, 1, 0, 0.25f);
  }
}
